// Attention_4870492913905
// MI455X (gfx1250) — matrix operand order repaired and checked by reading, not compiled, not run on hardware
//
#include <hip/hip_runtime.h>
#include <stdint.h>

// Problem constants (match reference)
#define Bn   2
#define Nn   320
#define DIMn 256
#define EDn  128
#define Hn   8
#define DHn  64
#define INn  512
#define NEG_MAX (-3.402823466e38f)

typedef __attribute__((ext_vector_type(16))) _Float16 v16h;
typedef __attribute__((ext_vector_type(8)))  float    v8f;

__device__ __forceinline__ uint16_t f2h(float f) {
  _Float16 h = (_Float16)f;
  return __builtin_bit_cast(uint16_t, h);
}
// pack two floats -> two f16 in one dword (lo = first)
__device__ __forceinline__ uint32_t pk2(float a, float b) {
  return (uint32_t)f2h(a) | ((uint32_t)f2h(b) << 16);
}

union HVec { v16h v; uint32_t u[8]; };

// A operand (M=16 x K=32, f16) gathered from row-major LDS (row stride ld elems).
// lanes 0-15: M=lane, K={0..7,16..23}; lanes 16-31: M=lane-16, K={8..15,24..31}.
__device__ __forceinline__ v16h wmma_load_A(const uint16_t* s, int ld, int lane) {
  HVec r;
  const uint16_t* row = s + (lane & 15) * ld + ((lane < 16) ? 0 : 8);
#pragma unroll
  for (int i = 0; i < 8; ++i) {
    int k = ((i < 4) ? 0 : 16) + (i & 3) * 2;
    r.u[i] = *(const uint32_t*)(row + k);
  }
  return r.v;
}

// B operand (K=32 x N=16, f16) from LDS stored transposed s[n*ld + k].
__device__ __forceinline__ v16h wmma_load_B(const uint16_t* s, int ld, int lane) {
  HVec r;
  const uint16_t* row = s + (lane & 15) * ld + ((lane < 16) ? 0 : 8);
#pragma unroll
  for (int i = 0; i < 8; ++i)
    r.u[i] = *(const uint32_t*)(row + ((i < 4) ? 0 : 16) + (i & 3) * 2);
  return r.v;
}

__device__ __forceinline__ v8f wmma_f16(v16h a, v16h b, v8f c) {
  return __builtin_amdgcn_wmma_f32_16x16x32_f16(false, a, false, b, (short)0, c,
                                                false, false);
}

// ---------------------------------------------------------------------------
// Generic C = A[M,K] @ W[K,N] + bias, f32 in/out, f16 WMMA compute.
// grid = (N/128, M/16), 256 threads (8 waves; wave w owns 16-col tile w).
// ---------------------------------------------------------------------------
__global__ __launch_bounds__(256) void gemm_bias_kernel(
    const float* __restrict__ A, int lda, const float* __restrict__ W, int ldw,
    const float* __restrict__ bias, float* __restrict__ C, int ldc, int K) {
  __shared__ __align__(16) uint16_t sA[16 * 32];
  __shared__ __align__(16) uint16_t sW[128 * 32];  // transposed [n][k]
  const int tid = threadIdx.x, lane = tid & 31, wave = tid >> 5;
  const int m0 = blockIdx.y * 16, n0 = blockIdx.x * 128;
  v8f acc = {0.f, 0.f, 0.f, 0.f, 0.f, 0.f, 0.f, 0.f};
  for (int kc = 0; kc < K; kc += 32) {
    // A tile: 16 rows x 8 float4
    if (tid < 16 * 8) {
      int r = tid >> 3, c4 = (tid & 7) * 4;
      float4 f = *(const float4*)(A + (size_t)(m0 + r) * lda + kc + c4);
      *(uint2*)(sA + r * 32 + c4) = make_uint2(pk2(f.x, f.y), pk2(f.z, f.w));
    }
    // W tile: 16 k-pairs x 32 float4 of n -> transposed pair-packed b32 stores
#pragma unroll
    for (int i = tid; i < 16 * 32; i += 256) {
      int k2 = i >> 5, n4 = (i & 31) * 4;
      const float* w0 = W + (size_t)(kc + 2 * k2) * ldw + n0 + n4;
      float4 fa = *(const float4*)w0;
      float4 fb = *(const float4*)(w0 + ldw);
      *(uint32_t*)(sW + (n4 + 0) * 32 + 2 * k2) = pk2(fa.x, fb.x);
      *(uint32_t*)(sW + (n4 + 1) * 32 + 2 * k2) = pk2(fa.y, fb.y);
      *(uint32_t*)(sW + (n4 + 2) * 32 + 2 * k2) = pk2(fa.z, fb.z);
      *(uint32_t*)(sW + (n4 + 3) * 32 + 2 * k2) = pk2(fa.w, fb.w);
    }
    __syncthreads();
    v16h a = wmma_load_A(sA, 32, lane);
    v16h b = wmma_load_B(sW + wave * 16 * 32, 32, lane);
    acc = wmma_f16(a, b, acc);
    __syncthreads();
  }
  int n = n0 + wave * 16 + (lane & 15);
  float bb = bias[n];
  int mhi = (lane < 16) ? 0 : 8;
#pragma unroll
  for (int r = 0; r < 8; ++r)
    C[(size_t)(m0 + r + mhi) * ldc + n] = acc[r] + bb;
}

// ---------------------------------------------------------------------------
// w~[b,i,h,k] = sum_d We[k, h*64+d] * q[b,i,h*64+d]   (stored f16, [bi][h][k])
// ---------------------------------------------------------------------------
__global__ __launch_bounds__(256) void wtilde_kernel(
    const float* __restrict__ q, const float* __restrict__ We,
    uint16_t* __restrict__ wt) {
  const int bi = blockIdx.x;
  __shared__ float sq[INn];
#pragma unroll
  for (int i = threadIdx.x; i < INn / 4; i += 256)
    *(float4*)(sq + i * 4) = *(const float4*)(q + (size_t)bi * INn + i * 4);
  __syncthreads();
#pragma unroll
  for (int idx = threadIdx.x; idx < Hn * EDn; idx += 256) {
    int h = idx >> 7, k = idx & 127;
    const float4* wr = (const float4*)(We + (size_t)k * INn + h * DHn);
    const float4* qr = (const float4*)(sq + h * DHn);
    float s = 0.f;
#pragma unroll
    for (int d = 0; d < DHn / 4; ++d) {
      float4 w = wr[d], x = qr[d];
      s = fmaf(w.x, x.x, s); s = fmaf(w.y, x.y, s);
      s = fmaf(w.z, x.z, s); s = fmaf(w.w, x.w, s);
    }
    wt[(size_t)bi * (Hn * EDn) + idx] = f2h(s);
  }
}

// ---------------------------------------------------------------------------
// simE[b,h,i,j] = sum_k edges[b,i,j,k] * w~[b,i,h,k]
// one block per (b,i); WMMA: M=j-tile(16), K=128 (4 chunks), N=16 (8 heads+pad)
// ---------------------------------------------------------------------------
__global__ __launch_bounds__(256) void sim_edge_kernel(
    const float* __restrict__ edges, const uint16_t* __restrict__ wt,
    float* __restrict__ sim) {
  const int bi = blockIdx.x, b = bi / Nn, i = bi % Nn;
  const int tid = threadIdx.x, lane = tid & 31, wave = tid >> 5;
  __shared__ __align__(16) uint16_t sWt[16 * EDn];    // [n=h(pad16)][k]
  __shared__ __align__(16) uint16_t sE[8][16 * EDn];  // per-wave [j16][k128]
#pragma unroll
  for (int idx = tid; idx < 16 * EDn; idx += 256) {
    int n = idx >> 7, k = idx & 127;
    sWt[idx] = (n < Hn) ? wt[(size_t)bi * (Hn * EDn) + n * EDn + k] : (uint16_t)0;
  }
  __syncthreads();
  v16h bop[4];
#pragma unroll
  for (int c = 0; c < 4; ++c) bop[c] = wmma_load_B(sWt + c * 32, EDn, lane);
  const float* ebase = edges + (size_t)bi * Nn * EDn;
  for (int jt = wave; jt < Nn / 16; jt += 8) {
    uint16_t* sAw = sE[wave];
    const float* esrc = ebase + (size_t)jt * 16 * EDn;
    // 16 rows x 32 float4, 32 lanes -> 16 float4 each, fully unrolled
#pragma unroll
    for (int idx = lane; idx < 16 * 32; idx += 32) {
      int r = idx >> 5, c4 = (idx & 31) * 4;
      float4 f = *(const float4*)(esrc + r * EDn + c4);
      *(uint2*)(sAw + r * EDn + c4) = make_uint2(pk2(f.x, f.y), pk2(f.z, f.w));
    }
    v8f acc = {0.f, 0.f, 0.f, 0.f, 0.f, 0.f, 0.f, 0.f};
#pragma unroll
    for (int c = 0; c < 4; ++c)
      acc = wmma_f16(wmma_load_A(sAw + c * 32, EDn, lane), bop[c], acc);
    int h = lane & 15;
    if (h < Hn) {
      int jbase = jt * 16 + ((lane < 16) ? 0 : 8);
#pragma unroll
      for (int r = 0; r < 8; ++r)
        sim[(((size_t)b * Hn + h) * Nn + i) * Nn + jbase + r] = acc[r];
    }
  }
}

// ---------------------------------------------------------------------------
// Attention core: one block per (b, h, 16-row i-tile).
// sim = (q.k^T + simE)*scale, masked; full-row softmax in LDS; out_v = P@V + be
// ---------------------------------------------------------------------------
__global__ __launch_bounds__(256) void attn_kernel(
    const float* __restrict__ qg, const float* __restrict__ kvg,
    const float* __restrict__ simE, const uint8_t* __restrict__ mask,
    const float* __restrict__ be, uint16_t* __restrict__ pg,
    float* __restrict__ voutg) {
  const int NT = Nn / 16;
  const int it = blockIdx.x % NT;
  const int h = (blockIdx.x / NT) % Hn;
  const int b = blockIdx.x / (NT * Hn);
  const int i0 = it * 16;
  const int tid = threadIdx.x, lane = tid & 31, wave = tid >> 5;

  __shared__ __align__(16) uint16_t sQ[16 * 64];      // [i][d]
  __shared__ __align__(16) uint16_t sKp[8][16 * 64];  // per-wave k tile [j][d]
  __shared__ __align__(16) uint16_t sVp[4][16 * 32];  // per-wave v tile [d][j]
  __shared__ __align__(16) uint16_t sP[16 * Nn];      // probs f16 [i][j]
  __shared__ __align__(16) float    sSim[16 * Nn];    // logits f32 [i][j]

  // q tile: 16 x 16 float4
  if (tid < 16 * 16) {
    int r = tid >> 4, c4 = (tid & 15) * 4;
    float4 f = *(const float4*)(qg + (size_t)(b * Nn + i0 + r) * INn + h * DHn + c4);
    *(uint2*)(sQ + r * 64 + c4) = make_uint2(pk2(f.x, f.y), pk2(f.z, f.w));
  }
  __syncthreads();

  const float scale = 0.125f;  // DH^-0.5
  const uint8_t* mrow = mask + b * Nn;
  v16h a0 = wmma_load_A(sQ, 64, lane);
  v16h a1 = wmma_load_A(sQ + 32, 64, lane);

  // ---- logits: q.k^T (WMMA) + simE, scale, mask -> sSim ----
  for (int t = wave; t < NT; t += 8) {
    uint16_t* sKw = sKp[wave];
    // 16 rows x 16 float4 over 32 lanes -> 8 float4 each
#pragma unroll
    for (int idx = lane; idx < 16 * 16; idx += 32) {
      int r = idx >> 4, c4 = (idx & 15) * 4;
      float4 f = *(const float4*)(kvg + (size_t)(b * Nn + t * 16 + r) * (2 * INn)
                                  + h * DHn + c4);
      *(uint2*)(sKw + r * 64 + c4) = make_uint2(pk2(f.x, f.y), pk2(f.z, f.w));
    }
    v8f acc = {0.f, 0.f, 0.f, 0.f, 0.f, 0.f, 0.f, 0.f};
    acc = wmma_f16(a0, wmma_load_B(sKw, 64, lane), acc);
    acc = wmma_f16(a1, wmma_load_B(sKw + 32, 64, lane), acc);
    int jl = lane & 15, j = t * 16 + jl;
    bool mj = mrow[j] != 0;
    int ihi = (lane < 16) ? 0 : 8;
#pragma unroll
    for (int r = 0; r < 8; ++r) {
      int il = r + ihi;
      float v = (acc[r] +
                 simE[(((size_t)b * Hn + h) * Nn + (i0 + il)) * Nn + j]) * scale;
      bool mi = mrow[i0 + il] != 0;
      sSim[il * Nn + j] = (mi && mj) ? v : NEG_MAX;
    }
  }
  __syncthreads();

  // ---- softmax: wave w owns rows 2w, 2w+1 ----
#pragma unroll
  for (int rr = 0; rr < 2; ++rr) {
    int row = wave * 2 + rr;
    float* srow = sSim + row * Nn;
    float m = NEG_MAX;
#pragma unroll
    for (int j = lane; j < Nn; j += 32) m = fmaxf(m, srow[j]);
#pragma unroll
    for (int off = 16; off > 0; off >>= 1) m = fmaxf(m, __shfl_xor(m, off, 32));
    float s = 0.f;
#pragma unroll
    for (int j = lane; j < Nn; j += 32) {
      float e = __expf(srow[j] - m);
      srow[j] = e;
      s += e;
    }
#pragma unroll
    for (int off = 16; off > 0; off >>= 1) s += __shfl_xor(s, off, 32);
    float inv = 1.f / s;
#pragma unroll
    for (int j = lane; j < Nn; j += 32) {
      uint16_t pb = f2h(srow[j] * inv);
      sP[row * Nn + j] = pb;
      pg[(((size_t)b * Hn + h) * Nn + (i0 + row)) * Nn + j] = pb;
    }
  }
  __syncthreads();

  // ---- out_v = P @ V : waves 0..3 each own a 16-wide d tile ----
  if (wave < 4) {
    uint16_t* sVw = sVp[wave];
    v8f acc = {0.f, 0.f, 0.f, 0.f, 0.f, 0.f, 0.f, 0.f};
    for (int kc = 0; kc < Nn; kc += 32) {
      // v tile transposed [d16][j32]: 16 j-pairs x 4 float4 of d
#pragma unroll
      for (int idx = lane; idx < 16 * 4; idx += 32) {
        int j2 = idx >> 2, d4 = (idx & 3) * 4;
        const float* v0 = kvg + (size_t)(b * Nn + kc + 2 * j2) * (2 * INn) + INn +
                          h * DHn + wave * 16 + d4;
        float4 fa = *(const float4*)v0;
        float4 fb = *(const float4*)(v0 + 2 * INn);
        *(uint32_t*)(sVw + (d4 + 0) * 32 + 2 * j2) = pk2(fa.x, fb.x);
        *(uint32_t*)(sVw + (d4 + 1) * 32 + 2 * j2) = pk2(fa.y, fb.y);
        *(uint32_t*)(sVw + (d4 + 2) * 32 + 2 * j2) = pk2(fa.z, fb.z);
        *(uint32_t*)(sVw + (d4 + 3) * 32 + 2 * j2) = pk2(fa.w, fb.w);
      }
      acc = wmma_f16(wmma_load_A(sP + kc, Nn, lane),
                     wmma_load_B(sVw, 32, lane), acc);
    }
    int d = wave * 16 + (lane & 15);
    float bias = be[h * DHn + d];  // sum_j p == 1 -> be added once
    int ihi = (lane < 16) ? 0 : 8;
#pragma unroll
    for (int r = 0; r < 8; ++r) {
      int il = r + ihi;
      voutg[(size_t)(b * Nn + i0 + il) * INn + h * DHn + d] = acc[r] + bias;
    }
  }
}

// ---------------------------------------------------------------------------
// g[h,k] = sum_j p[b,h,i,j]*edges[b,i,j,k]; vout[b,i,h*64+d] += g[h]@We_h
// one block per (b,i); wave w accumulates k-tile w over all j chunks.
// ---------------------------------------------------------------------------
__global__ __launch_bounds__(256) void edge_out_kernel(
    const float* __restrict__ edges, const uint16_t* __restrict__ pg,
    const float* __restrict__ We, float* __restrict__ voutg) {
  const int bi = blockIdx.x, b = bi / Nn, i = bi % Nn;
  const int tid = threadIdx.x, lane = tid & 31, wave = tid >> 5;
  __shared__ __align__(16) uint16_t sEt[EDn * 32];  // transposed [k][j]
  __shared__ __align__(16) uint16_t sPt[16 * 32];   // [h(pad16)][j]
  __shared__ __align__(16) float sG[Hn * EDn];
  v8f acc = {0.f, 0.f, 0.f, 0.f, 0.f, 0.f, 0.f, 0.f};
  const float* ebase = edges + (size_t)bi * Nn * EDn;
  for (int j0 = 0; j0 < Nn; j0 += 32) {
    // edges tile transposed: 16 j-pairs x 32 float4 of k
#pragma unroll
    for (int i2 = tid; i2 < 16 * 32; i2 += 256) {
      int j2 = i2 >> 5, k4 = (i2 & 31) * 4;
      const float* e0 = ebase + (size_t)(j0 + 2 * j2) * EDn + k4;
      float4 fa = *(const float4*)e0;
      float4 fb = *(const float4*)(e0 + EDn);
      *(uint32_t*)(sEt + (k4 + 0) * 32 + 2 * j2) = pk2(fa.x, fb.x);
      *(uint32_t*)(sEt + (k4 + 1) * 32 + 2 * j2) = pk2(fa.y, fb.y);
      *(uint32_t*)(sEt + (k4 + 2) * 32 + 2 * j2) = pk2(fa.z, fb.z);
      *(uint32_t*)(sEt + (k4 + 3) * 32 + 2 * j2) = pk2(fa.w, fb.w);
    }
    // probs tile [h(pad16)][32j], 4 probs per thread
    if (tid < 16 * 8) {
      int n = tid >> 3, j4 = (tid & 7) * 4;
      uint2 p = (n < Hn)
                    ? *(const uint2*)(pg + (((size_t)b * Hn + n) * Nn + i) * Nn +
                                      j0 + j4)
                    : make_uint2(0u, 0u);
      *(uint2*)(sPt + n * 32 + j4) = p;
    }
    __syncthreads();
    acc = wmma_f16(wmma_load_A(sPt, 32, lane),
                   wmma_load_B(sEt + (wave * 16) * 32, 32, lane), acc);
    __syncthreads();
  }
  {
    int k = wave * 16 + (lane & 15);
    int hhi = (lane < 16) ? 0 : 8;
#pragma unroll
    for (int r = 0; r < 8; ++r) {
      int hh = r + hhi;
      if (hh < Hn) sG[hh * EDn + k] = acc[r];
    }
  }
  __syncthreads();
#pragma unroll
  for (int idx = tid; idx < INn; idx += 256) {
    int h = idx >> 6, d = idx & 63;
    float s = 0.f;
    const float* gr = sG + h * EDn;
    const float* wc = We + h * DHn + d;
#pragma unroll 8
    for (int k = 0; k < EDn; ++k) s = fmaf(gr[k], wc[(size_t)k * INn], s);
    voutg[(size_t)bi * INn + idx] += s;  // block owns row bi: no race
  }
}

// ---------------------------------------------------------------------------
extern "C" void kernel_launch(void* const* d_in, const int* in_sizes, int n_in,
                              void* d_out, int out_size, void* d_ws,
                              size_t ws_size, hipStream_t stream) {
  const float*   nodes = (const float*)d_in[0];
  const float*   edges = (const float*)d_in[1];
  const uint8_t* mask  = (const uint8_t*)d_in[2];  // jax bool -> 1 byte/elem
  const float*   Wq    = (const float*)d_in[3];
  const float*   bq    = (const float*)d_in[4];
  const float*   Wkv   = (const float*)d_in[5];
  const float*   bkv   = (const float*)d_in[6];
  const float*   We    = (const float*)d_in[7];
  const float*   be    = (const float*)d_in[8];
  const float*   Wo    = (const float*)d_in[9];
  const float*   bo    = (const float*)d_in[10];
  float* out = (float*)d_out;

  char* ws = (char*)d_ws;
  float*    ws_q    = (float*)ws;    ws += (size_t)Bn * Nn * INn * 4;
  float*    ws_kv   = (float*)ws;    ws += (size_t)Bn * Nn * 2 * INn * 4;
  uint16_t* ws_wt   = (uint16_t*)ws; ws += (size_t)Bn * Nn * Hn * EDn * 2;
  float*    ws_sim  = (float*)ws;    ws += (size_t)Bn * Hn * Nn * Nn * 4;
  uint16_t* ws_p    = (uint16_t*)ws; ws += (size_t)Bn * Hn * Nn * Nn * 2;
  float*    ws_vout = (float*)ws;    ws += (size_t)Bn * Nn * INn * 4;
  (void)ws_size; (void)in_sizes; (void)n_in; (void)out_size;

  dim3 blk(256);
  // q = nodes@Wq + bq ; kv = nodes@Wkv + bkv
  gemm_bias_kernel<<<dim3(INn / 128, (Bn * Nn) / 16), blk, 0, stream>>>(
      nodes, DIMn, Wq, INn, bq, ws_q, INn, DIMn);
  gemm_bias_kernel<<<dim3((2 * INn) / 128, (Bn * Nn) / 16), blk, 0, stream>>>(
      nodes, DIMn, Wkv, 2 * INn, bkv, ws_kv, 2 * INn, DIMn);
  // w~ = We_h @ q_h  (per b,i,h)
  wtilde_kernel<<<dim3(Bn * Nn), blk, 0, stream>>>(ws_q, We, ws_wt);
  // simE = edges . w~
  sim_edge_kernel<<<dim3(Bn * Nn), blk, 0, stream>>>(edges, ws_wt, ws_sim);
  // attention: logits + softmax + P@V (+be)
  attn_kernel<<<dim3(Bn * Hn * (Nn / 16)), blk, 0, stream>>>(
      ws_q, ws_kv, ws_sim, mask, be, ws_p, ws_vout);
  // edge output path: g = P^T.edges ; vout += g@We
  edge_out_kernel<<<dim3(Bn * Nn), blk, 0, stream>>>(edges, ws_p, We, ws_vout);
  // out = vout@Wo + bo
  gemm_bias_kernel<<<dim3(DIMn / 128, (Bn * Nn) / 16), blk, 0, stream>>>(
      ws_vout, INn, Wo, DIMn, bo, out, DIMn, INn);
}
